// Controller_21449066676318
// MI455X (gfx1250) — matrix operand order repaired and checked by reading, not compiled, not run on hardware
//
#include <hip/hip_runtime.h>
#include <hip/hip_bf16.h>

#define H   256
#define PP  24
#define CC  32
#define BN  16384
#define G4  (4 * H)
#define WST 264   // LDS row stride in elements (256 + 8 pad -> 4-bank step per row)

typedef __attribute__((ext_vector_type(16))) __bf16 v16bf;
typedef __attribute__((ext_vector_type(8)))  __bf16 v8bf;
typedef __attribute__((ext_vector_type(8)))  float  v8f;

__device__ __forceinline__ float sigf(float x) { return 1.0f / (1.0f + __expf(-x)); }

// ---------------- weight fp32 -> bf16 conversion ----------------
__global__ void convert_weights(const float* __restrict__ wih, const float* __restrict__ whh,
                                __bf16* __restrict__ wih_b, __bf16* __restrict__ whh_b) {
    int i = blockIdx.x * blockDim.x + threadIdx.x;
    if (i < G4 * H) {
        wih_b[i] = (__bf16)wih[i];
        whh_b[i] = (__bf16)whh[i];
    }
}

// ---------------- step 0: rank-1 LSTM cell (h=c=0, x broadcast) ----------------
__global__ void init_cell(const float* __restrict__ start, const float* __restrict__ wih,
                          const float* __restrict__ bih, const float* __restrict__ bhh,
                          float* __restrict__ h1, float* __restrict__ c1) {
    int j = threadIdx.x;  // 0..255
    float gi = bih[j]         + bhh[j];
    float gf = bih[H + j]     + bhh[H + j];
    float gg = bih[2 * H + j] + bhh[2 * H + j];
    float go = bih[3 * H + j] + bhh[3 * H + j];
    for (int k = 0; k < H; ++k) {
        float x = start[k];
        gi += x * wih[(size_t)(j) * H + k];
        gf += x * wih[(size_t)(H + j) * H + k];
        gg += x * wih[(size_t)(2 * H + j) * H + k];
        go += x * wih[(size_t)(3 * H + j) * H + k];
    }
    (void)gf;  // f-gate multiplies c0 == 0
    float cc = sigf(gi) * tanhf(gg);
    float hh = sigf(go) * tanhf(cc);
    h1[j] = hh;
    c1[j] = cc;
}

__global__ void bcast_state(const float* __restrict__ h1, const float* __restrict__ c1,
                            float* __restrict__ hf, __bf16* __restrict__ hb,
                            float* __restrict__ cf) {
    int t = blockIdx.x * blockDim.x + threadIdx.x;
    if (t < BN * H) {
        int k = t & (H - 1);
        float hv = h1[k];
        hf[t] = hv;
        hb[t] = (__bf16)hv;
        cf[t] = c1[k];
    }
}

// ---------------- head: logits + gumbel-argmax + log-softmax + emb gather ----------------
// one wave32 per batch row; lane == choice index
__global__ __launch_bounds__(256) void head_sample(
    int p, const float* __restrict__ hf, const float* __restrict__ Whead,
    const float* __restrict__ bhead, const float* __restrict__ emb,
    const float* __restrict__ gumu, __bf16* __restrict__ xb,
    int* __restrict__ out_idx, float* __restrict__ lp_ws) {
    int lane = threadIdx.x;                                   // 0..31
    int b = blockIdx.x * blockDim.y + threadIdx.y;            // row
    const float* hrow = hf + (size_t)b * H;
    const float* wrow = Whead + ((size_t)p * CC + lane) * H;
    float acc = bhead[p * CC + lane];
#pragma unroll 4
    for (int k = 0; k < H; k += 4) {
        float4 hv = *(const float4*)(hrow + k);
        float4 wv = *(const float4*)(wrow + k);
        acc += hv.x * wv.x + hv.y * wv.y + hv.z * wv.z + hv.w * wv.w;
    }
    float u = gumu[(((size_t)p * BN) + b) * CC + lane];
    float g = -__logf(-__logf(u + 1e-9f) + 1e-9f);

    // gumbel-max argmax (first-index tie-break to match jnp.argmax)
    float kv = acc + g;
    int ki = lane;
#pragma unroll
    for (int off = 16; off > 0; off >>= 1) {
        float ov = __shfl_xor(kv, off, 32);
        int oi = __shfl_xor(ki, off, 32);
        if (ov > kv || (ov == kv && oi < ki)) { kv = ov; ki = oi; }
    }
    // log-softmax
    float m = acc;
#pragma unroll
    for (int off = 16; off > 0; off >>= 1) m = fmaxf(m, __shfl_xor(m, off, 32));
    float s = __expf(acc - m);
#pragma unroll
    for (int off = 16; off > 0; off >>= 1) s += __shfl_xor(s, off, 32);
    float lsel = __shfl(acc, ki, 32);
    if (lane == 0) {
        out_idx[(size_t)b * PP + p] = ki;
        lp_ws[(size_t)p * BN + b] = lsel - m - __logf(s);
    }
    // x = emb[p][ki]  (store bf16 for the WMMA A-operand)
    const float* erow = emb + ((size_t)p * CC + ki) * H;
#pragma unroll
    for (int t = 0; t < 8; ++t) {
        int k = lane * 8 + t;
        xb[(size_t)b * H + k] = (__bf16)erow[k];
    }
}

// ---------------- fused LSTM gates GEMM (bf16 WMMA, LDS-staged weights) ----------------
// block = 8 waves sharing one j-tile (weights staged in LDS), each wave owns TWO
// 16-row M-tiles (32 batch rows) so every LDS B-tile read feeds two WMMAs.
// Per k-slice: 2 A chunks + 4 B tiles loaded into distinct regs, then 8 WMMAs.
__global__ __launch_bounds__(256) void lstm_gates(
    const __bf16* __restrict__ xb, const __bf16* __restrict__ hb_in,
    const __bf16* __restrict__ wih, const __bf16* __restrict__ whh,
    const float* __restrict__ bih, const float* __restrict__ bhh,
    float* __restrict__ cf, float* __restrict__ hf, __bf16* __restrict__ hb_out) {
    __shared__ __bf16 lw[4 * 16 * WST];  // one matrix's j-tile slice: ~33 KB

    int lane = threadIdx.x;                       // 0..31
    int tid = threadIdx.y * 32 + lane;            // 0..255
    int jtile = blockIdx.x & 15;                  // shared by all 8 waves
    int mbase = ((blockIdx.x >> 4) * 8 + threadIdx.y) * 32;  // 32 rows per wave
    int jbase = jtile * 16;
    int lo = lane & 15, hi = lane >> 4;

    v8f acc[2][4];
#pragma unroll
    for (int gq = 0; gq < 4; ++gq) {
        float bias = bih[gq * H + jbase + lo] + bhh[gq * H + jbase + lo];
#pragma unroll
        for (int r = 0; r < 8; ++r) { acc[0][gq][r] = bias; acc[1][gq][r] = bias; }
    }

    // A-operand layout: lane -> M = lane&15, K-chunk offset = (lane>>4)*8 (+16 hi half)
    const int koff = hi * 8;
    const int kb = hi * 16;  // unused; B-operand K base is koff, same as A
    const __bf16* Xrow0 = xb + (size_t)(mbase + lo) * H;
    const __bf16* Xrow1 = xb + (size_t)(mbase + 16 + lo) * H;
    const __bf16* Hrow0 = hb_in + (size_t)(mbase + lo) * H;
    const __bf16* Hrow1 = hb_in + (size_t)(mbase + 16 + lo) * H;

#pragma unroll
    for (int phase = 0; phase < 2; ++phase) {
        const __bf16* W = phase ? whh : wih;
        const __bf16* Arow0 = phase ? Hrow0 : Xrow0;
        const __bf16* Arow1 = phase ? Hrow1 : Xrow1;

        __syncthreads();  // previous phase's LDS reads complete
        // stage 64 rows x 256 K of this j-tile's weights: 2048 16B chunks / 256 thr
#pragma unroll
        for (int c = tid; c < 2048; c += 256) {
            int row64 = c >> 5;       // gq*16 + n
            int k8 = c & 31;          // K/8
            int gq = row64 >> 4, n = row64 & 15;
            *(v8bf*)&lw[row64 * WST + k8 * 8] =
                *(const v8bf*)(W + (size_t)(gq * H + jbase + n) * H + k8 * 8);
        }
        __syncthreads();

#pragma unroll
        for (int k0 = 0; k0 < H; k0 += 32) {
            // A chunks for both M-tiles
            v8bf a00 = *(const v8bf*)(Arow0 + k0 + koff);
            v8bf a01 = *(const v8bf*)(Arow0 + k0 + koff + 16);
            v8bf a10 = *(const v8bf*)(Arow1 + k0 + koff);
            v8bf a11 = *(const v8bf*)(Arow1 + k0 + koff + 16);
            v16bf a[2];
#pragma unroll
            for (int i = 0; i < 8; ++i) {
                a[0][i] = a00[i]; a[0][i + 8] = a01[i];
                a[1][i] = a10[i]; a[1][i + 8] = a11[i];
            }
            // all four gate B-tiles up front (independent regs -> clause + ILP)
            v16bf bb[4];
#pragma unroll
            for (int gq = 0; gq < 4; ++gq) {
                const __bf16* wl = &lw[(gq * 16 + lo) * WST + k0 + koff];
                v8bf b0 = *(const v8bf*)(wl);
                v8bf b1 = *(const v8bf*)(wl + 16);
#pragma unroll
                for (int i = 0; i < 8; ++i) { bb[gq][i] = b0[i]; bb[gq][i + 8] = b1[i]; }
            }
            // 8 independent accumulation chains
#pragma unroll
            for (int gq = 0; gq < 4; ++gq) {
                acc[0][gq] = __builtin_amdgcn_wmma_f32_16x16x32_bf16(
                    false, a[0], false, bb[gq], (short)0, acc[0][gq], false, false);
                acc[1][gq] = __builtin_amdgcn_wmma_f32_16x16x32_bf16(
                    false, a[1], false, bb[gq], (short)0, acc[1][gq], false, false);
            }
        }
    }

    // pointwise LSTM cell: element r of tile m2 -> row mbase + 16*m2 + r + 8*hi
#pragma unroll
    for (int m2 = 0; m2 < 2; ++m2) {
#pragma unroll
        for (int r = 0; r < 8; ++r) {
            size_t off = (size_t)(mbase + 16 * m2 + r + 8 * hi) * H + (jbase + lo);
            float iv = acc[m2][0][r], fv = acc[m2][1][r];
            float gv = acc[m2][2][r], ov = acc[m2][3][r];
            float c2 = sigf(fv) * cf[off] + sigf(iv) * tanhf(gv);
            float h2 = sigf(ov) * tanhf(c2);
            cf[off] = c2;
            hf[off] = h2;
            hb_out[off] = (__bf16)h2;
        }
    }
}

// ---------------- final: sum per-step log-probs ----------------
__global__ void sum_lp(const float* __restrict__ lp_ws, float* __restrict__ out_lp) {
    int b = blockIdx.x * blockDim.x + threadIdx.x;
    if (b < BN) {
        float s = 0.0f;
        for (int p = 0; p < PP; ++p) s += lp_ws[(size_t)p * BN + b];
        out_lp[b] = s;
    }
}

extern "C" void kernel_launch(void* const* d_in, const int* in_sizes, int n_in,
                              void* d_out, int out_size, void* d_ws, size_t ws_size,
                              hipStream_t stream) {
    (void)in_sizes; (void)n_in; (void)out_size; (void)ws_size;
    const float* start = (const float*)d_in[0];
    const float* Wih   = (const float*)d_in[1];
    const float* Whh   = (const float*)d_in[2];
    const float* bih   = (const float*)d_in[3];
    const float* bhh   = (const float*)d_in[4];
    const float* Whead = (const float*)d_in[5];
    const float* bhead = (const float*)d_in[6];
    const float* emb   = (const float*)d_in[7];
    const float* gumu  = (const float*)d_in[8];

    char* ws = (char*)d_ws;
    size_t off = 0;
    auto alloc = [&](size_t bytes) -> char* {
        char* q = ws + off;
        off += (bytes + 255) & ~(size_t)255;
        return q;
    };
    __bf16* wih_b = (__bf16*)alloc((size_t)G4 * H * 2);
    __bf16* whh_b = (__bf16*)alloc((size_t)G4 * H * 2);
    __bf16* hb0   = (__bf16*)alloc((size_t)BN * H * 2);
    __bf16* hb1   = (__bf16*)alloc((size_t)BN * H * 2);
    __bf16* xb    = (__bf16*)alloc((size_t)BN * H * 2);
    float*  hf    = (float*)alloc((size_t)BN * H * 4);
    float*  cf    = (float*)alloc((size_t)BN * H * 4);
    float*  lpw   = (float*)alloc((size_t)PP * BN * 4);
    float*  h1    = (float*)alloc(H * 4);
    float*  c1    = (float*)alloc(H * 4);

    int*   out_idx = (int*)d_out;
    float* out_lp  = (float*)d_out + (size_t)BN * PP;

    convert_weights<<<(G4 * H + 255) / 256, 256, 0, stream>>>(Wih, Whh, wih_b, whh_b);
    init_cell<<<1, 256, 0, stream>>>(start, Wih, bih, bhh, h1, c1);
    bcast_state<<<(BN * H + 255) / 256, 256, 0, stream>>>(h1, c1, hf, hb0, cf);

    dim3 blk(32, 8);  // 8 wave32s per block
    for (int p = 0; p < PP; ++p) {
        head_sample<<<BN / 8, blk, 0, stream>>>(p, hf, Whead, bhead, emb, gumu, xb,
                                                out_idx, lpw);
        __bf16* hin  = (p & 1) ? hb1 : hb0;
        __bf16* hout = (p & 1) ? hb0 : hb1;
        // blocks = 16 j-tiles * 64 m-groups (8 waves x 32 rows each) = 1024
        lstm_gates<<<16 * (BN / 32 / 8), blk, 0, stream>>>(
            xb, hin, wih_b, whh_b, bih, bhh, cf, hf, hout);
    }
    sum_lp<<<BN / 256, 256, 0, stream>>>(lpw, out_lp);
}
